// NetworkForAGraphWithNodeAttributes_25821343384349
// MI455X (gfx1250) — compile-verified, matrix operand order checked by reading, not run on hardware
//
#include <hip/hip_runtime.h>
#include <hip/hip_bf16.h>
#include <math.h>

typedef __attribute__((ext_vector_type(16))) _Float16 v16h;
typedef __attribute__((ext_vector_type(2)))  _Float16 v2h;
typedef __attribute__((ext_vector_type(8)))  float    v8f;

#define N_NODES   50000
#define N_EDGES   800000
#define D_ATTR    16
#define N_BASIS   10
#define FC_HIDDEN 100
#define SH_DIM    9
#define MAX_RADIUS 3.5f

// ---------------------------------------------------------------------------
// Edge geometry: spherical harmonics (lmax=2) + radial bump embedding
// ---------------------------------------------------------------------------
__global__ void edge_geom_kernel(const float* __restrict__ pos,
                                 const int* __restrict__ eidx,
                                 float* __restrict__ Y,
                                 float* __restrict__ emb,
                                 int nE) {
    int e = blockIdx.x * blockDim.x + threadIdx.x;
    if (e >= nE) return;
    int s = eidx[e], d = eidx[nE + e];
    float vx = pos[s*3+0] - pos[d*3+0];
    float vy = pos[s*3+1] - pos[d*3+1];
    float vz = pos[s*3+2] - pos[d*3+2];
    float r  = sqrtf(vx*vx + vy*vy + vz*vz);
    float inv = 1.0f / fmaxf(r, 1e-12f);
    float x = vx*inv, y = vy*inv, z = vz*inv;
    const float s3 = 1.7320508075688772f, s5 = 2.23606797749979f, s15 = 3.872983346207417f;
    float* Ye = Y + (size_t)e * SH_DIM;
    Ye[0] = 1.0f;
    Ye[1] = s3*x;  Ye[2] = s3*y;  Ye[3] = s3*z;
    Ye[4] = s15*x*y;  Ye[5] = s15*y*z;
    Ye[6] = 0.5f*s5*(3.0f*z*z - 1.0f);
    Ye[7] = s15*x*z;  Ye[8] = 0.5f*s15*(x*x - y*y);

    const float step = MAX_RADIUS / (float)(N_BASIS + 1);
    float* Ee = emb + (size_t)e * N_BASIS;
#pragma unroll
    for (int b = 0; b < N_BASIS; ++b) {
        float c  = MAX_RADIUS * (float)(b + 1) / (float)(N_BASIS + 1);
        float dd = (r - c) / step;
        float v  = 0.0f;
        if (fabsf(dd) < 1.0f) {
            float den = 1.0f - dd*dd;
            v = 1.14136f * __expf(2.0f - 1.0f/den);
        }
        Ee[b] = v * 3.1622776601683795f;   // * sqrt(N_BASIS)
    }
}

// ---------------------------------------------------------------------------
// CSR-by-destination build (edge topology is layer-invariant: built once)
// ---------------------------------------------------------------------------
__global__ void zero_int_kernel(int* __restrict__ p, int n) {
    int i = blockIdx.x * blockDim.x + threadIdx.x;
    if (i < n) p[i] = 0;
}

__global__ void count_deg_kernel(const int* __restrict__ eidx,
                                 int* __restrict__ deg, int nE) {
    int e = blockIdx.x * blockDim.x + threadIdx.x;
    if (e >= nE) return;
    atomicAdd(&deg[eidx[nE + e]], 1);
}

// single-block exclusive scan of deg -> rowstart (nN up to ~64k)
__global__ void __launch_bounds__(256)
scan_kernel(const int* __restrict__ deg, int* __restrict__ rowstart, int nN) {
    __shared__ int partial[256];
    int t = threadIdx.x;
    int ch = (nN + 255) / 256;
    int beg = t * ch;
    int end = beg + ch; if (end > nN) end = nN;
    int s = 0;
    for (int i = beg; i < end; ++i) s += deg[i];
    partial[t] = s;
    __syncthreads();
    for (int off = 1; off < 256; off <<= 1) {
        int v = partial[t];
        int u = (t >= off) ? partial[t - off] : 0;
        __syncthreads();
        partial[t] = v + u;
        __syncthreads();
    }
    int run = (t == 0) ? 0 : partial[t - 1];
    for (int i = beg; i < end; ++i) {
        rowstart[i] = run;
        run += deg[i];
    }
    if (t == 255) rowstart[nN] = run;
}

__global__ void fill_csr_kernel(const int* __restrict__ eidx,
                                const int* __restrict__ rowstart,
                                int* __restrict__ cursor,
                                int* __restrict__ csr_eid,
                                int* __restrict__ csr_src, int nE) {
    int e = blockIdx.x * blockDim.x + threadIdx.x;
    if (e >= nE) return;
    int dN = eidx[nE + e];
    int p  = rowstart[dN] + atomicAdd(&cursor[dN], 1);
    csr_eid[p] = e;
    csr_src[p] = eidx[e];
}

// ---------------------------------------------------------------------------
// Pack weight tensor (64,16,dout) f32 into WMMA-fragment-major f16 layout:
//   P[((ks*nt + t)*32 + lane)*16 + idx]  with idx=2v+e2 per the ISA 16-bit
//   B-matrix layout -> each lane's fragment is one contiguous 32B load.
// ---------------------------------------------------------------------------
__global__ void pack_w_kernel(const float* __restrict__ W,
                              _Float16* __restrict__ P, int dout) {
    int nt = dout >> 4;
    int total = 32 * nt * 32 * 16;       // ksteps * ntiles * lanes * elems
    int t = blockIdx.x * blockDim.x + threadIdx.x;
    if (t >= total) return;
    int idx  = t & 15;
    int lane = (t >> 4) & 31;
    int rem  = t >> 9;
    int tt   = rem % nt;
    int ks   = rem / nt;
    int v  = idx >> 1, e2 = idx & 1;
    int g  = lane >> 4;
    int kb = (v < 4) ? (2*v + 8*g) : (16 + 2*(v-4) + 8*g);
    int k  = ks*32 + kb + e2;
    int col = tt*16 + (lane & 15);
    P[t] = (_Float16)W[(size_t)k*dout + col];
}

__global__ void zero_kernel(float* __restrict__ p, int n) {
    int i = blockIdx.x * blockDim.x + threadIdx.x;
    if (i < n) p[i] = 0.0f;
}

// ---------------------------------------------------------------------------
// WMMA bilinear: out[n,k] = sum_{i,j} x[n,i]*attr[n,j]*W[i,j,k] * scale
// GEMM Z(50000 x 1024) @ Wflat(1024 x dout); Z built on the fly in registers
// from f16 LDS tiles with v_pk_mul_f16. One wave = 16 rows. NT0/NT1 are the
// number of 16-column output tiles for the two (shared-A) weight tensors.
// ---------------------------------------------------------------------------
template<int NT0, int NT1>
__global__ void __launch_bounds__(128)
bilinear_wmma_kernel(const float* __restrict__ x,
                     const float* __restrict__ attr,
                     const _Float16* __restrict__ W0p,
                     const _Float16* __restrict__ W1p,
                     float* __restrict__ out0,
                     float* __restrict__ out1,
                     int nTiles, float scale) {
    __shared__ _Float16 xs [4][16][64];
    __shared__ _Float16 ats[4][16][16];

    int w    = threadIdx.x >> 5;
    int lane = threadIdx.x & 31;
    int tile = blockIdx.x * 4 + w;
    if (tile >= nTiles) return;
    int m0 = tile * 16;

    // stage this wave's 16-node x/attr tile into LDS as f16 (same-wave use)
    for (int idx = lane; idx < 16*64; idx += 32) {
        int r = idx >> 6, c = idx & 63;
        xs[w][r][c] = (_Float16)x[(size_t)(m0 + r)*64 + c];
    }
    for (int idx = lane; idx < 16*16; idx += 32) {
        int r = idx >> 4, c = idx & 15;
        ats[w][r][c] = (_Float16)attr[(size_t)(m0 + r)*16 + c];
    }

    int m = lane & 15;        // row within tile (A); col within tile (B/C/D)
    int g = lane >> 4;

    v8f acc0[NT0] = {};
    v8f acc1[(NT1 > 0) ? NT1 : 1] = {};

    const _Float16* p0 = W0p + (size_t)lane * 16;
    const _Float16* p1 = W1p + (size_t)lane * 16;

    for (int ks = 0; ks < 32; ++ks) {         // K = 1024 in steps of 32
        // A fragment 16x32 f16 (ISA layout): pairs (2v,2v+1) share x[m][i],
        // use adjacent attr values -> 8x v_pk_mul_f16
        int i0 = ks * 2;
        _Float16 xi0 = xs[w][m][i0];
        _Float16 xi1 = xs[w][m][i0 + 1];
        v2h x0v = {xi0, xi0};
        v2h x1v = {xi1, xi1};
        union { v16h v; v2h h2[8]; } A;
#pragma unroll
        for (int p = 0; p < 4; ++p) {
            v2h ap = *(const v2h*)&ats[w][m][8*g + 2*p];
            A.h2[p]     = x0v * ap;
            A.h2[4 + p] = x1v * ap;
        }
#pragma unroll
        for (int t = 0; t < NT0; ++t) {
            v16h b = *(const v16h*)(p0 + ((size_t)(ks*NT0 + t) * 32) * 16);
            acc0[t] = __builtin_amdgcn_wmma_f32_16x16x32_f16(
                false, A.v, false, b, (short)0, acc0[t], false, false);
        }
        if constexpr (NT1 > 0) {
#pragma unroll
            for (int t = 0; t < NT1; ++t) {
                v16h b = *(const v16h*)(p1 + ((size_t)(ks*NT1 + t) * 32) * 16);
                acc1[t] = __builtin_amdgcn_wmma_f32_16x16x32_f16(
                    false, A.v, false, b, (short)0, acc1[t], false, false);
            }
        }
    }

    // C/D layout: VGPR r holds M = r + 8*g, N = lane&15
    const int dout0 = NT0 * 16;
#pragma unroll
    for (int t = 0; t < NT0; ++t) {
#pragma unroll
        for (int r = 0; r < 8; ++r) {
            int row = m0 + r + 8*g;
            out0[(size_t)row*dout0 + t*16 + m] = acc0[t][r] * scale;
        }
    }
    if constexpr (NT1 > 0) {
        const int dout1 = NT1 * 16;
#pragma unroll
        for (int t = 0; t < NT1; ++t) {
#pragma unroll
            for (int r = 0; r < 8; ++r) {
                int row = m0 + r + 8*g;
                out1[(size_t)row*dout1 + t*16 + m] = acc1[t][r] * scale;
            }
        }
    }
}

// ---------------------------------------------------------------------------
// Per-edge radial MLP: s[e] = dot(silu(emb@fc1)@fc2, Y[e]) / sqrt(16)
// ---------------------------------------------------------------------------
__global__ void __launch_bounds__(256)
edge_s_kernel(const float* __restrict__ emb, const float* __restrict__ Y,
              const float* __restrict__ fc1, const float* __restrict__ fc2,
              float* __restrict__ sval, int nE) {
    __shared__ float f1[N_BASIS * FC_HIDDEN];
    __shared__ float f2[FC_HIDDEN * SH_DIM];
    for (int i = threadIdx.x; i < N_BASIS*FC_HIDDEN; i += 256) f1[i] = fc1[i];
    for (int i = threadIdx.x; i < FC_HIDDEN*SH_DIM;  i += 256) f2[i] = fc2[i];
    __syncthreads();

    int e = blockIdx.x * 256 + threadIdx.x;
    if (e >= nE) return;
    float em[N_BASIS];
#pragma unroll
    for (int i = 0; i < N_BASIS; ++i) em[i] = emb[(size_t)e*N_BASIS + i];
    float wv[SH_DIM];
#pragma unroll
    for (int k = 0; k < SH_DIM; ++k) wv[k] = 0.0f;
    for (int j = 0; j < FC_HIDDEN; ++j) {
        float a = 0.0f;
#pragma unroll
        for (int i = 0; i < N_BASIS; ++i) a += em[i] * f1[i*FC_HIDDEN + j];
        float sg = a / (1.0f + __expf(-a));           // silu
#pragma unroll
        for (int k = 0; k < SH_DIM; ++k) wv[k] += sg * f2[j*SH_DIM + k];
    }
    float s = 0.0f;
#pragma unroll
    for (int k = 0; k < SH_DIM; ++k) s += wv[k] * Y[(size_t)e*SH_DIM + k];
    sval[e] = s * 0.25f;                              // fold 1/sqrt(NUM_NEIGHBORS)
}

// ---------------------------------------------------------------------------
// CSR gather aggregation (atomic-free): one wave per node, 2 features/lane.
// agg[n,:] = sum_{e: dst[e]==n} h[src[e],:] * s[e]
// ---------------------------------------------------------------------------
__global__ void __launch_bounds__(256)
gather_kernel(const float* __restrict__ h,
              const float* __restrict__ sval,
              const int* __restrict__ rowstart,
              const int* __restrict__ csr_eid,
              const int* __restrict__ csr_src,
              float* __restrict__ agg, int nN) {
    int node = blockIdx.x * 8 + (threadIdx.x >> 5);
    int lane = threadIdx.x & 31;
    if (node >= nN) return;
    int beg = rowstart[node];
    int end = rowstart[node + 1];
    float a0 = 0.0f, a1 = 0.0f;
    for (int p = beg; p < end; ++p) {
        int e    = csr_eid[p];
        int srcN = csr_src[p];
        float s  = sval[e];
        const float* hr = h + (size_t)srcN * 64;
        a0 += hr[lane]      * s;
        a1 += hr[lane + 32] * s;
    }
    agg[(size_t)node*64 + lane]      = a0;
    agg[(size_t)node*64 + lane + 32] = a1;
}

// ---------------------------------------------------------------------------
// alpha[n] = sum_{i,j} agg[n,i]*attr[n,j]*Wa[i,j] / 32
// ---------------------------------------------------------------------------
__global__ void alpha_kernel(const float* __restrict__ agg,
                             const float* __restrict__ attr,
                             const float* __restrict__ Wa,
                             float* __restrict__ alpha, int nN) {
    int n = blockIdx.x * blockDim.x + threadIdx.x;
    if (n >= nN) return;
    float at[D_ATTR];
#pragma unroll
    for (int j = 0; j < D_ATTR; ++j) at[j] = attr[(size_t)n*D_ATTR + j];
    float acc = 0.0f;
    for (int i = 0; i < 64; ++i) {
        float ai = agg[(size_t)n*64 + i];
        float d = 0.0f;
#pragma unroll
        for (int j = 0; j < D_ATTR; ++j) d += at[j] * Wa[i*D_ATTR + j];
        acc += ai * d;
    }
    alpha[n] = acc * (1.0f / 32.0f);
}

// ---------------------------------------------------------------------------
// out = sc + alpha*conv, optional SiLU gate
// ---------------------------------------------------------------------------
__global__ void combine_kernel(const float* __restrict__ sc,
                               const float* __restrict__ conv,
                               const float* __restrict__ alpha,
                               float* __restrict__ xo,
                               int nN, int dout, int gate) {
    int t = blockIdx.x * blockDim.x + threadIdx.x;
    if (t >= nN * dout) return;
    int n = t / dout;
    float v = sc[t] + alpha[n] * conv[t];
    if (gate) v = v / (1.0f + __expf(-v));
    xo[t] = v;
}

// ---------------------------------------------------------------------------
// Graph pooling: out[batch[n], f] += x[n, f] / sqrt(N_NODES)
// ---------------------------------------------------------------------------
__global__ void pool_kernel(const float* __restrict__ xf,
                            const int* __restrict__ batch,
                            float* __restrict__ out, int nN) {
    int t = blockIdx.x * blockDim.x + threadIdx.x;
    if (t >= nN * 32) return;
    int n = t >> 5, f = t & 31;
    atomicAdd(&out[batch[n]*32 + f], xf[t] * 0.004472135955f);  // 1/sqrt(50000)
}

// ---------------------------------------------------------------------------
extern "C" void kernel_launch(void* const* d_in, const int* in_sizes, int n_in,
                              void* d_out, int out_size, void* d_ws, size_t ws_size,
                              hipStream_t stream) {
    const float* pos        = (const float*)d_in[0];
    const float* node_input = (const float*)d_in[1];
    const float* node_attr  = (const float*)d_in[2];
    const int*   eidx       = (const int*)d_in[3];
    const int*   batch      = (const int*)d_in[4];

    float* f = (float*)d_ws;
    size_t o = 0;
    float* Y     = f + o; o += (size_t)N_EDGES * SH_DIM;
    float* emb   = f + o; o += (size_t)N_EDGES * N_BASIS;
    float* sval  = f + o; o += (size_t)N_EDGES;
    float* xa    = f + o; o += (size_t)N_NODES * 64;
    float* xb    = f + o; o += (size_t)N_NODES * 64;
    float* sc    = f + o; o += (size_t)N_NODES * 64;
    float* hb    = f + o; o += (size_t)N_NODES * 64;
    float* agg   = f + o; o += (size_t)N_NODES * 64;
    float* conv  = f + o; o += (size_t)N_NODES * 64;
    float* alpha = f + o; o += (size_t)N_NODES;          // 50000 = 3125*16
    _Float16* wsc16 = (_Float16*)(f + o); o += 32768;    // 65536 halves, 32B-aligned
    _Float16* wl116 = (_Float16*)(f + o); o += 32768;
    _Float16* wl216 = (_Float16*)(f + o); o += 32768;
    int* deg      = (int*)(f + o); o += N_NODES;
    int* cursor   = (int*)(f + o); o += N_NODES;
    int* rowstart = (int*)(f + o); o += N_NODES + 16;    // nN+1, padded
    int* csr_eid  = (int*)(f + o); o += N_EDGES;
    int* csr_src  = (int*)(f + o); o += N_EDGES;

    // ---- layer-invariant precompute: edge geometry + CSR by destination ----
    edge_geom_kernel<<<(N_EDGES + 255)/256, 256, 0, stream>>>(pos, eidx, Y, emb, N_EDGES);

    zero_int_kernel<<<(N_NODES + 255)/256, 256, 0, stream>>>(deg, N_NODES);
    zero_int_kernel<<<(N_NODES + 255)/256, 256, 0, stream>>>(cursor, N_NODES);
    count_deg_kernel<<<(N_EDGES + 255)/256, 256, 0, stream>>>(eidx, deg, N_EDGES);
    scan_kernel<<<1, 256, 0, stream>>>(deg, rowstart, N_NODES);
    fill_csr_kernel<<<(N_EDGES + 255)/256, 256, 0, stream>>>(
        eidx, rowstart, cursor, csr_eid, csr_src, N_EDGES);

    const int douts[4] = {64, 64, 64, 32};
    const float* xcur = node_input;
    float* bufs[2] = {xa, xb};
    const int nTiles = N_NODES / 16;   // 3125, exact
    const int gBil = (nTiles + 3) / 4;
    const float bscale = 1.0f / 32.0f; // 1/sqrt(64*16)

    for (int l = 0; l < 4; ++l) {
        const float* Wsc = (const float*)d_in[5 + 6*l + 0];
        const float* Wl1 = (const float*)d_in[5 + 6*l + 1];
        const float* fc1 = (const float*)d_in[5 + 6*l + 2];
        const float* fc2 = (const float*)d_in[5 + 6*l + 3];
        const float* Wl2 = (const float*)d_in[5 + 6*l + 4];
        const float* Wa  = (const float*)d_in[5 + 6*l + 5];
        int dout = douts[l];
        int nP   = 32 * (dout >> 4) * 32 * 16;   // packed element count
        int nP64 = 32 * 4 * 32 * 16;

        pack_w_kernel<<<(nP   + 255)/256, 256, 0, stream>>>(Wsc, wsc16, dout);
        pack_w_kernel<<<(nP64 + 255)/256, 256, 0, stream>>>(Wl1, wl116, 64);
        pack_w_kernel<<<(nP   + 255)/256, 256, 0, stream>>>(Wl2, wl216, dout);

        // sc = bilinear(x, attr, W_sc); h = bilinear(x, attr, W_lin1)
        if (dout == 64)
            bilinear_wmma_kernel<4,4><<<gBil, 128, 0, stream>>>(
                xcur, node_attr, wsc16, wl116, sc, hb, nTiles, bscale);
        else
            bilinear_wmma_kernel<2,4><<<gBil, 128, 0, stream>>>(
                xcur, node_attr, wsc16, wl116, sc, hb, nTiles, bscale);

        edge_s_kernel<<<(N_EDGES + 255)/256, 256, 0, stream>>>(emb, Y, fc1, fc2, sval, N_EDGES);

        // atomic-free aggregation via CSR gather (one wave per node)
        gather_kernel<<<(N_NODES + 7)/8, 256, 0, stream>>>(
            hb, sval, rowstart, csr_eid, csr_src, agg, N_NODES);

        // conv = bilinear(agg, attr, W_lin2)
        if (dout == 64)
            bilinear_wmma_kernel<4,0><<<gBil, 128, 0, stream>>>(
                agg, node_attr, wl216, (const _Float16*)nullptr, conv, (float*)nullptr,
                nTiles, bscale);
        else
            bilinear_wmma_kernel<2,0><<<gBil, 128, 0, stream>>>(
                agg, node_attr, wl216, (const _Float16*)nullptr, conv, (float*)nullptr,
                nTiles, bscale);

        alpha_kernel<<<(N_NODES + 255)/256, 256, 0, stream>>>(agg, node_attr, Wa, alpha, N_NODES);

        float* xnext = bufs[l & 1];
        combine_kernel<<<(N_NODES*dout + 255)/256, 256, 0, stream>>>(
            sc, conv, alpha, xnext, N_NODES, dout, (l < 3) ? 1 : 0);
        xcur = xnext;
    }

    zero_kernel<<<(2048 + 255)/256, 256, 0, stream>>>((float*)d_out, 2048);
    pool_kernel<<<(N_NODES*32 + 255)/256, 256, 0, stream>>>(xcur, batch, (float*)d_out, N_NODES);
}
